// CachedTransformer_57930518888566
// MI455X (gfx1250) — matrix operand order repaired and checked by reading, not compiled, not run on hardware
//
#include <hip/hip_runtime.h>

// GPT prefill for MI455X (gfx1250, wave32). All matmuls via v_wmma_f32_16x16x32_bf16.
// bf16 operands everywhere (weights transposed+converted once); softmax reductions
// are single fused v_max_num_f32_dpp / v_add_f32_dpp instructions (no canon, no LDS).

namespace {

constexpr int Bc = 2, Sc = 1024, Vc = 32000, Dc = 1024, Hc = 16, Lc = 4;
constexpr int DKc = Dc / Hc;   // 64
constexpr int DFc = 4 * Dc;    // 4096
constexpr int Tok = Bc * Sc;   // 2048 rows

typedef __bf16 bf16_t;
typedef bf16_t   v16bf __attribute__((ext_vector_type(16)));
typedef float    v8f   __attribute__((ext_vector_type(8)));
typedef unsigned u32x4 __attribute__((ext_vector_type(4)));

union FragU { v16bf v; unsigned u[8]; };

__device__ __forceinline__ unsigned short bfbits(float f) {
  union { bf16_t h; unsigned short s; } c;
  c.h = (bf16_t)f;                               // hardware cvt
  return c.s;
}
__device__ __forceinline__ v8f wmma_bf16(v16bf a, v16bf b, v8f c) {
  return __builtin_amdgcn_wmma_f32_16x16x32_bf16(false, a, false, b, (short)0, c,
                                                 false, false);
}

// Raw max (inputs are never sNaN -> skip llvm.maxnum canonicalization)
__device__ __forceinline__ float fmax_raw(float a, float b) {
  float d;
  asm("v_max_num_f32_e32 %0, %1, %2" : "=v"(d) : "v"(a), "v"(b));
  return d;
}

// ---- fused DPP row reductions (within 16-lane groups; all lanes get result) ----
__device__ __forceinline__ float rowmax16(float x) {
  float t;
  asm("v_max_num_f32_dpp %0, %1, %1 quad_perm:[1,0,3,2] row_mask:0xf bank_mask:0xf bound_ctrl:1"
      : "=v"(t) : "v"(x)); x = t;
  asm("v_max_num_f32_dpp %0, %1, %1 quad_perm:[2,3,0,1] row_mask:0xf bank_mask:0xf bound_ctrl:1"
      : "=v"(t) : "v"(x)); x = t;
  asm("v_max_num_f32_dpp %0, %1, %1 row_half_mirror row_mask:0xf bank_mask:0xf bound_ctrl:1"
      : "=v"(t) : "v"(x)); x = t;
  asm("v_max_num_f32_dpp %0, %1, %1 row_mirror row_mask:0xf bank_mask:0xf bound_ctrl:1"
      : "=v"(t) : "v"(x)); x = t;
  return x;
}
__device__ __forceinline__ float rowsum16(float x) {
  float t;
  asm("v_add_f32_dpp %0, %1, %1 quad_perm:[1,0,3,2] row_mask:0xf bank_mask:0xf bound_ctrl:1"
      : "=v"(t) : "v"(x)); x = t;
  asm("v_add_f32_dpp %0, %1, %1 quad_perm:[2,3,0,1] row_mask:0xf bank_mask:0xf bound_ctrl:1"
      : "=v"(t) : "v"(x)); x = t;
  asm("v_add_f32_dpp %0, %1, %1 row_half_mirror row_mask:0xf bank_mask:0xf bound_ctrl:1"
      : "=v"(t) : "v"(x)); x = t;
  asm("v_add_f32_dpp %0, %1, %1 row_mirror row_mask:0xf bank_mask:0xf bound_ctrl:1"
      : "=v"(t) : "v"(x)); x = t;
  return x;
}

// ---------------------------------------------------------------- embedding
__global__ void embed_ker(const int* __restrict__ ids,
                          const float* __restrict__ emb, float* __restrict__ x) {
  const int row = blockIdx.x, t = threadIdx.x;
  const float* e = emb + (size_t)ids[row] * Dc;
  float* xr = x + (size_t)row * Dc;
#pragma unroll
  for (int i = 0; i < Dc / 256; ++i) xr[t + 256 * i] = e[t + 256 * i];
}

// ------------------------------------------- weight transpose + fp32->bf16
// W[K,N] fp32 -> WT[N,K] bf16 bits. grid (N/32, K/32), 256 threads.
__global__ void convT_ker(const float* __restrict__ W, unsigned short* __restrict__ WT,
                          int K, int N) {
  __shared__ float tile[32][33];
  const int n0 = blockIdx.x * 32, k0 = blockIdx.y * 32;
  const int tx = threadIdx.x & 31, ty = threadIdx.x >> 5;   // 8 rows/step
#pragma unroll
  for (int r = ty; r < 32; r += 8) tile[r][tx] = W[(size_t)(k0 + r) * N + (n0 + tx)];
  __syncthreads();
#pragma unroll
  for (int r = ty; r < 32; r += 8)
    WT[(size_t)(n0 + r) * K + (k0 + tx)] = bfbits(tile[tx][r]);
}

// --------------------------------------------------- layernorm (bf16 out)
__global__ void ln_ker(const float* __restrict__ in, const float* __restrict__ g,
                       const float* __restrict__ be, unsigned short* __restrict__ out) {
  const int row = blockIdx.x, t = threadIdx.x;
  const float* p = in + (size_t)row * Dc;
  float v4[Dc / 256], s = 0.f, s2 = 0.f;
#pragma unroll
  for (int i = 0; i < Dc / 256; ++i) {
    float f = p[t + 256 * i];
    v4[i] = f; s += f; s2 += f * f;
  }
#pragma unroll
  for (int off = 16; off >= 1; off >>= 1) {
    s  += __shfl_xor(s,  off, 32);
    s2 += __shfl_xor(s2, off, 32);
  }
  __shared__ float rs[8], rs2[8];
  const int lane = t & 31, wid = t >> 5;
  if (lane == 0) { rs[wid] = s; rs2[wid] = s2; }
  __syncthreads();
  float ts = 0.f, ts2 = 0.f;
#pragma unroll
  for (int j = 0; j < 8; ++j) { ts += rs[j]; ts2 += rs2[j]; }
  const float mean = ts * (1.f / Dc);
  const float var  = ts2 * (1.f / Dc) - mean * mean;
  const float r    = rsqrtf(var + 1e-5f);
  unsigned short* o = out + (size_t)row * Dc;
#pragma unroll
  for (int i = 0; i < Dc / 256; ++i) {
    const int col = t + 256 * i;
    o[col] = bfbits((v4[i] - mean) * r * g[col] + be[col]);
  }
}

// ---------------------------------------------------------------- WMMA GEMM
// C[M,N] = act( A[M,K] * W[K,N] + bias ) (+resid). A: bf16 [M,K]; WT: bf16 [N,K].
// OUTM: 0 = fp32 row-major (+resid), 1 = bf16 row-major (*oscale),
//       2 = bf16 V-transposed [B,H,DK,S]
constexpr int TM = 128, TN = 128, TKs = 32;
constexpr int LW = (TKs + 2) / 2;   // 17 u32 words per LDS row (pad, odd stride)

template <bool BIAS, bool ACT_GELU, bool RESID, int OUTM>
__global__ __launch_bounds__(256) void gemm_bf16_wmma(
    const unsigned* __restrict__ A2, const unsigned* __restrict__ WT2,
    const float* __restrict__ bias, const float* resid, void* outp,
    int M, int N, int K, float oscale) {
  __shared__ unsigned lA[TM][LW];   // bf16 pairs, row-major [m][k/2]
  __shared__ unsigned lB[TN][LW];   // bf16 pairs, transposed [n][k/2]

  const int t = threadIdx.x;
  const int lane = t & 31, wid = t >> 5;
  const int wm = wid & 3, wn = wid >> 2;          // 4 waves on M, 2 on N
  const int bm = blockIdx.y * TM, bn = blockIdx.x * TN;
  const int half = lane >> 4, l16 = lane & 15;
  const int kbA = half * 8;                       // A frag: interleaved halves
  const int kbB = half * 16;                      // unused; B frag reads kbA order like A
  const int KW = K >> 1;                          // u32 words per global row

  const v8f vzero = {0.f, 0.f, 0.f, 0.f, 0.f, 0.f, 0.f, 0.f};
  v8f acc[2][4];
#pragma unroll
  for (int i = 0; i < 2; ++i)
#pragma unroll
    for (int j = 0; j < 4; ++j) acc[i][j] = vzero;

  // staging: each thread handles half a row (8 words) of each tile via 2x b128
  const int sr = t >> 1, sw = (t & 1) * 8;
  const unsigned* aRow = A2  + (size_t)(bm + sr) * KW + sw;
  const unsigned* bRow = WT2 + (size_t)(bn + sr) * KW + sw;

  u32x4 pa0, pa1, pb0, pb1;
  auto ldRegs = [&](int kw) __attribute__((always_inline)) {
    pa0 = *(const u32x4*)(aRow + kw);
    pa1 = *(const u32x4*)(aRow + kw + 4);
    pb0 = *(const u32x4*)(bRow + kw);
    pb1 = *(const u32x4*)(bRow + kw + 4);
  };
  auto stLDS = [&]() __attribute__((always_inline)) {
#pragma unroll
    for (int i = 0; i < 4; ++i) {
      lA[sr][sw + i] = pa0[i]; lA[sr][sw + 4 + i] = pa1[i];
      lB[sr][sw + i] = pb0[i]; lB[sr][sw + 4 + i] = pb1[i];
    }
  };
  ldRegs(0);
  stLDS();

  for (int k0 = 0; k0 < K; k0 += TKs) {
    __syncthreads();
    const bool more = (k0 + TKs) < K;
    if (more) ldRegs((k0 + TKs) >> 1);    // pipeline: next tile during WMMA

    v16bf aF[2], bF[4];
#pragma unroll
    for (int i = 0; i < 2; ++i) {
      const unsigned short* rp = (const unsigned short*)&lA[32 * wm + 16 * i + l16][0];
      FragU u;
#pragma unroll
      for (int p = 0; p < 4; ++p) {
        u.u[p]     = *(const unsigned*)(rp + kbA + 2 * p);
        u.u[4 + p] = *(const unsigned*)(rp + kbA + 16 + 2 * p);
      }
      aF[i] = u.v;
    }
#pragma unroll
    for (int j = 0; j < 4; ++j) {
      const unsigned short* cp = (const unsigned short*)&lB[64 * wn + 16 * j + l16][0];
      FragU u;
#pragma unroll
      for (int r = 0; r < 8; ++r) u.u[r] = *(const unsigned*)(cp + kbA + 16 * (r >> 2) + 2 * (r & 3));
      bF[j] = u.v;
    }
#pragma unroll
    for (int i = 0; i < 2; ++i)
#pragma unroll
      for (int j = 0; j < 4; ++j) acc[i][j] = wmma_bf16(aF[i], bF[j], acc[i][j]);
    __syncthreads();
    if (more) stLDS();
  }

#pragma unroll
  for (int i = 0; i < 2; ++i)
#pragma unroll
    for (int j = 0; j < 4; ++j) {
      const int col = bn + 64 * wn + 16 * j + l16;
      const float bc = BIAS ? bias[col] : 0.f;
#pragma unroll
      for (int v = 0; v < 8; ++v) {
        const int row = bm + 32 * wm + 16 * i + v + 8 * half;
        float x = acc[i][j][v];
        if (BIAS) x += bc;
        if (ACT_GELU) x = 0.5f * x * (1.f + erff(x * 0.70710678118654752f));
        if (OUTM == 0) {
          const size_t o = (size_t)row * N + col;
          if (RESID) x += resid[o];
          ((float*)outp)[o] = x;
        } else if (OUTM == 1) {
          ((unsigned short*)outp)[(size_t)row * N + col] = bfbits(x * oscale);
        } else {  // bf16 [B,H,DK,S]
          const int b = row >> 10, s = row & (Sc - 1);
          const int h = col >> 6, d = col & (DKc - 1);
          ((unsigned short*)outp)[((size_t)(b * Hc + h) * DKc + d) * Sc + s] = bfbits(x);
        }
      }
    }
}

// ------------------------------------------------------- fused flash attention
// grid (S/64, B*H); 128 thr = 4 waves; wave owns 16 q-rows; 32 keys per step.
// Q (pre-scaled), K: bf16 [B,S,D]; V: bf16 [B,H,DK,S]. Output: bf16 [B,S,D].
constexpr int PADP = 2;

__global__ __launch_bounds__(128) void attn_flash_wmma(
    const unsigned short* __restrict__ Qp, const unsigned short* __restrict__ Kp,
    const unsigned short* __restrict__ VTp, unsigned short* __restrict__ Op) {
  __shared__ unsigned short ldsP[4][16][32 + PADP];  // wave-private P bounce
  const int lane = threadIdx.x & 31, w = threadIdx.x >> 5;
  const int half = lane >> 4, l16 = lane & 15;
  const int bh = blockIdx.y, b = bh >> 4, h = bh & (Hc - 1);
  const int q0 = blockIdx.x * 64 + w * 16;
  const size_t base  = ((size_t)b * Sc) * Dc + (size_t)h * DKc;    // Q,K rows
  const size_t vbase = (size_t)(b * Hc + h) * DKc * Sc;            // V^T rows
  const int kbA = half * 8, kbB = half * 16;

  v16bf qf[2];
  {
    const unsigned short* qr = Qp + base + (size_t)(q0 + l16) * Dc;
#pragma unroll
    for (int f = 0; f < 2; ++f) {
      FragU u;
#pragma unroll
      for (int p = 0; p < 4; ++p) {
        const int d0 = 32 * f + kbA + 2 * p;
        u.u[p]     = *(const unsigned*)(qr + d0);
        u.u[4 + p] = *(const unsigned*)(qr + d0 + 16);
      }
      qf[f] = u.v;
    }
  }

  const v8f vzero = {0.f, 0.f, 0.f, 0.f, 0.f, 0.f, 0.f, 0.f};
  float m[8], lsum[8];   // lsum is LANE-LOCAL partial; reduced once at the end
  v8f o[4];
#pragma unroll
  for (int v = 0; v < 8; ++v) { m[v] = -3.0e38f; lsum[v] = 0.f; }
#pragma unroll
  for (int j = 0; j < 4; ++j) o[j] = vzero;

  auto kstep = [&](int c0, bool domask) __attribute__((always_inline)) {
    float p[2][8];
#pragma unroll
    for (int g = 0; g < 2; ++g) {          // scores: Q(16x64) . K^T -> 16x16
      const unsigned short* kr = Kp + base + (size_t)(c0 + 16 * g + l16) * Dc;
      v8f s = vzero;
#pragma unroll
      for (int f = 0; f < 2; ++f) {
        FragU u;
#pragma unroll
        for (int r = 0; r < 8; ++r)
          u.u[r] = *(const unsigned*)(kr + 32 * f + kbA + 16 * (r >> 2) + 2 * (r & 3));
        s = wmma_bf16(qf[f], u.v, s);
      }
#pragma unroll
      for (int v = 0; v < 8; ++v) {
        if (domask) {
          const int col = c0 + 16 * g + l16;
          const int row = q0 + v + 8 * half;
          p[g][v] = (col > row) ? -3.0e38f : s[v];
        } else {
          p[g][v] = s[v];
        }
      }
    }
    // online softmax: fused DPP row-max (group-uniform); lane-local sums
#pragma unroll
    for (int v = 0; v < 8; ++v) {
      const float t = rowmax16(fmax_raw(p[0][v], p[1][v]));
      const float mn = fmax_raw(m[v], t);
      const float c = __expf(m[v] - mn);
      m[v] = mn;
      const float p0 = __expf(p[0][v] - mn), p1 = __expf(p[1][v] - mn);
      p[0][v] = p0; p[1][v] = p1;
      lsum[v] = lsum[v] * c + (p0 + p1);
#pragma unroll
      for (int j = 0; j < 4; ++j) o[j][v] *= c;
    }
    // bounce P through LDS: C-layout -> A-layout bf16 (intra-wave hazard: dscnt)
#pragma unroll
    for (int g = 0; g < 2; ++g)
#pragma unroll
      for (int v = 0; v < 8; ++v)
        ldsP[w][v + 8 * half][16 * g + l16] = bfbits(p[g][v]);
    asm volatile("s_wait_dscnt 0" ::: "memory");
    v16bf pf;
    {
      const unsigned short* pr = &ldsP[w][l16][0];
      FragU u;
#pragma unroll
      for (int q = 0; q < 4; ++q) {
        u.u[q]     = *(const unsigned*)(pr + kbA + 2 * q);
        u.u[4 + q] = *(const unsigned*)(pr + kbA + 16 + 2 * q);
      }
      pf = u.v;
    }
    asm volatile("" ::: "memory");
    // O += P(16x32) x V(32x64): V^T rows contiguous in kv -> b128 loads
#pragma unroll
    for (int j = 0; j < 4; ++j) {
      const unsigned short* vc = VTp + vbase + (size_t)(16 * j + l16) * Sc;
      FragU u;
#pragma unroll
      for (int r = 0; r < 8; ++r) u.u[r] = *(const unsigned*)(vc + c0 + kbA + 16 * (r >> 2) + 2 * (r & 3));
      o[j] = wmma_bf16(pf, u.v, o[j]);
    }
  };

  const int ktMax  = (q0 + 15) >> 5;                           // inclusive
  const int ktFull = (q0 >= 31) ? (((q0 - 31) >> 5) + 1) : 0;  // mask-free blocks
  for (int kt = 0; kt < ktFull; ++kt) kstep(kt * 32, false);
  for (int kt = ktFull; kt <= ktMax; ++kt) kstep(kt * 32, true);

  // final cross-lane sum, normalize, store bf16
#pragma unroll
  for (int v = 0; v < 8; ++v) {
    const float inv = 1.0f / rowsum16(lsum[v]);
    const int row = q0 + v + 8 * half;
#pragma unroll
    for (int j = 0; j < 4; ++j)
      Op[base + (size_t)row * Dc + 16 * j + l16] = bfbits(o[j][v] * inv);
  }
}

}  // namespace

extern "C" void kernel_launch(void* const* d_in, const int* in_sizes, int n_in,
                              void* d_out, int out_size, void* d_ws, size_t ws_size,
                              hipStream_t stream) {
  const int*   ids  = (const int*)  d_in[0];
  const float* emb  = (const float*)d_in[1];
  const float* Wq   = (const float*)d_in[2];
  const float* Wk   = (const float*)d_in[3];
  const float* Wv   = (const float*)d_in[4];
  const float* Wo   = (const float*)d_in[5];
  const float* W1   = (const float*)d_in[6];
  const float* b1   = (const float*)d_in[7];
  const float* W2   = (const float*)d_in[8];
  const float* b2   = (const float*)d_in[9];
  const float* ln1g = (const float*)d_in[10];
  const float* ln1b = (const float*)d_in[11];
  const float* ln2g = (const float*)d_in[12];
  const float* ln2b = (const float*)d_in[13];
  const float* lnfg = (const float*)d_in[14];
  const float* lnfb = (const float*)d_in[15];
  const float* hw   = (const float*)d_in[16];

  // workspace (~130 MB): fp32 x + bf16 activations + bf16 transposed weights
  float* x = (float*)d_ws;
  unsigned short* hb   = (unsigned short*)(x + (size_t)Tok * Dc);
  unsigned short* att  = hb  + (size_t)Tok * Dc;
  unsigned short* mid  = att + (size_t)Tok * Dc;
  unsigned short* qb   = mid + (size_t)Tok * DFc;
  unsigned short* kbuf = qb  + (size_t)Tok * Dc;
  unsigned short* vT   = kbuf + (size_t)Tok * Dc;
  unsigned short* wqT  = vT  + (size_t)Tok * Dc;
  unsigned short* wkT  = wqT + (size_t)Dc * Dc;
  unsigned short* wvT  = wkT + (size_t)Dc * Dc;
  unsigned short* woT  = wvT + (size_t)Dc * Dc;
  unsigned short* w1T  = woT + (size_t)Dc * Dc;
  unsigned short* w2T  = w1T + (size_t)Dc * DFc;
  unsigned short* whT  = w2T + (size_t)Dc * DFc;

  const dim3 gDD(Dc / TN, Tok / TM);    // (8,16)
  const dim3 gDF(DFc / TN, Tok / TM);   // (32,16)
  const dim3 gV (Vc / TN, Tok / TM);    // (250,16)
  const dim3 gAt(Sc / 64, Bc * Hc);     // (16,32)

  embed_ker<<<Tok, 256, 0, stream>>>(ids, emb, x);
  convT_ker<<<dim3(Vc / 32, Dc / 32), 256, 0, stream>>>(hw, whT, Dc, Vc);
  for (int l = 0; l < Lc; ++l) {
    const size_t oDD = (size_t)l * Dc * Dc, oDF = (size_t)l * Dc * DFc;
    convT_ker<<<dim3(Dc / 32, Dc / 32), 256, 0, stream>>>(Wq + oDD, wqT, Dc, Dc);
    convT_ker<<<dim3(Dc / 32, Dc / 32), 256, 0, stream>>>(Wk + oDD, wkT, Dc, Dc);
    convT_ker<<<dim3(Dc / 32, Dc / 32), 256, 0, stream>>>(Wv + oDD, wvT, Dc, Dc);
    convT_ker<<<dim3(Dc / 32, Dc / 32), 256, 0, stream>>>(Wo + oDD, woT, Dc, Dc);
    convT_ker<<<dim3(DFc / 32, Dc / 32), 256, 0, stream>>>(W1 + oDF, w1T, Dc, DFc);
    convT_ker<<<dim3(Dc / 32, DFc / 32), 256, 0, stream>>>(W2 + oDF, w2T, DFc, Dc);

    ln_ker<<<Tok, 256, 0, stream>>>(x, ln1g + l * Dc, ln1b + l * Dc, hb);
    gemm_bf16_wmma<false, false, false, 1><<<gDD, 256, 0, stream>>>(
        (const unsigned*)hb, (const unsigned*)wqT, nullptr, nullptr, qb,
        Tok, Dc, Dc, 0.125f);                       // Q pre-scaled by 1/sqrt(DK)
    gemm_bf16_wmma<false, false, false, 1><<<gDD, 256, 0, stream>>>(
        (const unsigned*)hb, (const unsigned*)wkT, nullptr, nullptr, kbuf,
        Tok, Dc, Dc, 1.0f);
    gemm_bf16_wmma<false, false, false, 2><<<gDD, 256, 0, stream>>>(
        (const unsigned*)hb, (const unsigned*)wvT, nullptr, nullptr, vT,
        Tok, Dc, Dc, 1.0f);
    attn_flash_wmma<<<gAt, 128, 0, stream>>>(qb, kbuf, vT, att);
    gemm_bf16_wmma<false, false, true, 0><<<gDD, 256, 0, stream>>>(
        (const unsigned*)att, (const unsigned*)woT, nullptr, x, x,
        Tok, Dc, Dc, 1.0f);
    ln_ker<<<Tok, 256, 0, stream>>>(x, ln2g + l * Dc, ln2b + l * Dc, hb);
    gemm_bf16_wmma<true, true, false, 1><<<gDF, 256, 0, stream>>>(
        (const unsigned*)hb, (const unsigned*)w1T, b1 + (size_t)l * DFc, nullptr,
        mid, Tok, DFc, Dc, 1.0f);
    gemm_bf16_wmma<true, false, true, 0><<<gDD, 256, 0, stream>>>(
        (const unsigned*)mid, (const unsigned*)w2T, b2 + (size_t)l * Dc, x, x,
        Tok, Dc, DFc, 1.0f);
  }
  ln_ker<<<Tok, 256, 0, stream>>>(x, lnfg, lnfb, hb);
  gemm_bf16_wmma<false, false, false, 0><<<gV, 256, 0, stream>>>(
      (const unsigned*)hb, (const unsigned*)whT, nullptr, nullptr, (float*)d_out,
      Tok, Vc, Dc, 1.0f);

  (void)in_sizes; (void)n_in; (void)out_size; (void)ws_size;
}
